// FingerPrinter_61486751809804
// MI455X (gfx1250) — compile-verified, matrix operand order checked by reading, not run on hardware
//
#include <hip/hip_runtime.h>
#include <hip/hip_bf16.h>

typedef _Float16 h16;
typedef __attribute__((ext_vector_type(8)))  _Float16 v8h;
typedef __attribute__((ext_vector_type(16))) _Float16 v16h;
typedef __attribute__((ext_vector_type(8)))  float    v8f;

#define FP_EPS 1e-3f

__device__ __forceinline__ float elu1(float v) { return v > 0.f ? v : __expf(v) - 1.f; }

// ---------------------------------------------------------------------------
// prep: per-conv weight transpose ([3][Cin][Cout] f32 -> [3][Cout][Cin] f16)
// and BN fold: scale = gamma*rsqrt(var+eps), shift = beta - mean*scale
// ---------------------------------------------------------------------------
__global__ void fp_prep_conv(const float* __restrict__ w,
                             const float* __restrict__ gamma,
                             const float* __restrict__ beta,
                             const float* __restrict__ mean,
                             const float* __restrict__ var,
                             h16* __restrict__ wT,
                             float* __restrict__ scale,
                             float* __restrict__ shift,
                             int Cin, int Cout, int n)
{
    int idx = blockIdx.x * 256 + threadIdx.x;
    if (idx < n) {
        int t  = idx / (Cin * Cout);
        int r  = idx - t * Cin * Cout;
        int co = r / Cin;
        int ci = r - co * Cin;
        wT[idx] = (h16)w[(t * Cin + ci) * Cout + co];
    }
    if (idx < Cout) {
        float sc   = gamma[idx] * rsqrtf(var[idx] + FP_EPS);
        scale[idx] = sc;
        shift[idx] = beta[idx] - mean[idx] * sc;
    }
}

// ---------------------------------------------------------------------------
// layer 0 conv1: (256,256,32,1) f32 -> (256,256,32,64) f16. K=3, memory-bound.
// ---------------------------------------------------------------------------
__global__ void fp_conv0(const float* __restrict__ x,
                         const h16*  __restrict__ wT,   // [3][64]
                         const float* __restrict__ bias,
                         const float* __restrict__ scale,
                         const float* __restrict__ shift,
                         h16* __restrict__ out)
{
    __shared__ float sw[3 * 64];
    __shared__ float sb[64], ssc[64], ssh[64];
    int tid = threadIdx.x;
    if (tid < 192) sw[tid] = (float)wT[tid];
    if (tid < 64) { sb[tid] = bias[tid]; ssc[tid] = scale[tid]; ssh[tid] = shift[tid]; }
    __syncthreads();

    int pos = blockIdx.x * 256 + tid;          // (b*256 + h)*32 + w, C=1
    int w   = pos & 31;
    float xm = (w > 0)  ? x[pos - 1] : 0.f;    // SAME pad left = 1
    float xc = x[pos];
    float xp = (w < 31) ? x[pos + 1] : 0.f;

    unsigned int* outp = (unsigned int*)(out + (size_t)pos * 64);
#pragma unroll
    for (int c = 0; c < 64; c += 2) {
        float v0 = xm * sw[c]     + xc * sw[64 + c]     + xp * sw[128 + c]     + sb[c];
        float v1 = xm * sw[c + 1] + xc * sw[64 + c + 1] + xp * sw[128 + c + 1] + sb[c + 1];
        v0 = elu1(v0) * ssc[c]     + ssh[c];
        v1 = elu1(v1) * ssc[c + 1] + ssh[c + 1];
        union { h16 h[2]; unsigned int u; } pk;
        pk.h[0] = (h16)v0; pk.h[1] = (h16)v1;
        outp[c >> 1] = pk.u;
    }
}

// ---------------------------------------------------------------------------
// Generic 3-tap conv as implicit GEMM with WMMA f16->f32.
//   out[b,p,l,co] = sum_{t,ci} in[b,p,(s*l + t - padL),ci] * wT[t][co][ci]
// M = B*P*Lout (m = (b*P+p)*Lout + l), N = Cout, K = 3*Cin (32-wide chunks).
// Block: 256 thr = 8 waves (4 M x 2 N). Wave tile: 16M x (16*NSUB)N.
// LDS double-buffered, regs prefetch 2 chunks ahead -> 1 barrier per chunk,
// one A fragment feeds NSUB v_wmma_f32_16x16x32_f16 per chunk.
// ---------------------------------------------------------------------------
template<int NSUB>
__global__ void __launch_bounds__(256) fp_conv_wmma(
    const h16*  __restrict__ in,
    const h16*  __restrict__ wT,     // [3][Cout][Cin] f16
    const float* __restrict__ bias,
    const float* __restrict__ scale,
    const float* __restrict__ shift,
    h16* __restrict__ out,
    int Cin, int Cout, int Lin, int Lout, int P,
    int s, int padL,
    int inXs, int inPs, int inBs,
    int outXs, int outPs, int outBs)
{
    constexpr int NT   = 32 * NSUB;   // block N tile (2 N-waves * 16*NSUB)
    constexpr int BROW = NSUB / 2;    // B rows staged per thread
    // row pitch 40 halfs (80B): 16B aligned, conflict-free b128 LDS reads
    __shared__ __align__(16) h16 sA[2][64 * 40];
    __shared__ __align__(16) h16 sB[2][NT * 40];

    const int tid  = threadIdx.x;
    const int lane = tid & 31;
    const int wave = tid >> 5;
    const int wm   = wave & 3;
    const int wn   = wave >> 2;

    const int m0   = blockIdx.x * 64;
    const int nblk = blockIdx.y * NT;

    // staging assignment: rows x 4 segments of 8 halfs
    const int row  = tid >> 2;
    const int seg  = tid & 3;
    const int segh = seg << 3;

    const int m    = m0 + row;
    const int l    = m % Lout;
    const int rem  = m / Lout;
    const int pp   = rem % P;
    const int bb   = rem / P;
    const int aBase = bb * inBs + pp * inPs + segh;
    const int x0    = l * s - padL;

    const int woc = Cout * Cin;
    const int cpt = Cin >> 5;         // K chunks per tap (Cin multiple of 32)
    const int nk  = 3 * cpt;

    v8f acc[NSUB] = {};

    // register staging (prefetch pipeline)
    v8h rA;
    v8h rB[BROW];
    int tS = 0, ccS = 0;              // (tap, chunk-in-tap) of next chunk to gload

    auto gload = [&]() {
        const int ci0 = ccS << 5;
        const int x   = x0 + tS;
        v8h v = {};
        if (x >= 0 && x < Lin) v = *(const v8h*)(in + aBase + x * inXs + ci0);
        rA = v;
        const int wb = tS * woc + ci0 + segh;
#pragma unroll
        for (int r = 0; r < BROW; ++r)
            rB[r] = *(const v8h*)(wT + wb + (nblk + r * 64 + row) * Cin);
        if (++ccS == cpt) { ccS = 0; ++tS; }
    };
    auto stage = [&](int buf) {
        *(v8h*)&sA[buf][row * 40 + segh] = rA;
#pragma unroll
        for (int r = 0; r < BROW; ++r)
            *(v8h*)&sB[buf][(r * 64 + row) * 40 + segh] = rB[r];
    };

    // prologue: chunk0 -> LDS buf0; chunk1 -> regs
    gload();
    stage(0);
    if (nk > 1) gload();

    const int kb = (lane >> 4) << 3;            // frag K offset: 0 or 8
    const int ra = wm * 16 + (lane & 15);
    int cur = 0;
    for (int kc = 0; kc < nk; ++kc) {
        __syncthreads();   // buf[cur] staged by all; buf[cur^1] readers done
        if (kc + 1 < nk) {
            stage(cur ^ 1);
            if (kc + 2 < nk) gload();
        }
        // fragment per ISA 16-bit A layout:
        // lane<16: K 0-7 / 16-23 of row M=lane ; lane>=16: K 8-15 / 24-31
        v8h alo = *(const v8h*)&sA[cur][ra * 40 + kb];
        v8h ahi = *(const v8h*)&sA[cur][ra * 40 + kb + 16];
        v16h afrag = __builtin_shufflevector(alo, ahi,
            0, 1, 2, 3, 4, 5, 6, 7, 8, 9, 10, 11, 12, 13, 14, 15);
#pragma unroll
        for (int sub = 0; sub < NSUB; ++sub) {
            const int rb = wn * (16 * NSUB) + sub * 16 + (lane & 15);
            v8h blo = *(const v8h*)&sB[cur][rb * 40 + kb];
            v8h bhi = *(const v8h*)&sB[cur][rb * 40 + kb + 16];
            v16h bfrag = __builtin_shufflevector(blo, bhi,
                0, 1, 2, 3, 4, 5, 6, 7, 8, 9, 10, 11, 12, 13, 14, 15);
            acc[sub] = __builtin_amdgcn_wmma_f32_16x16x32_f16(
                false, afrag, false, bfrag, (short)0, acc[sub], false, false);
        }
        cur ^= 1;
    }

    // epilogue: bias + ELU + folded BN, store f16
    float bi[NSUB], sc[NSUB], sh[NSUB];
    int   nn[NSUB];
#pragma unroll
    for (int sub = 0; sub < NSUB; ++sub) {
        nn[sub] = nblk + wn * (16 * NSUB) + sub * 16 + (lane & 15);
        bi[sub] = bias[nn[sub]];
        sc[sub] = scale[nn[sub]];
        sh[sub] = shift[nn[sub]];
    }
    const int mBase = m0 + wm * 16 + ((lane >> 4) << 3);
#pragma unroll
    for (int i = 0; i < 8; ++i) {
        const int mi  = mBase + i;
        const int li  = mi % Lout;
        const int re  = mi / Lout;
        const int pi  = re % P;
        const int bo  = re / P;
        const int ob  = bo * outBs + pi * outPs + li * outXs;
#pragma unroll
        for (int sub = 0; sub < NSUB; ++sub) {
            float v = acc[sub][i] + bi[sub];
            v = elu1(v) * sc[sub] + sh[sub];
            out[ob + nn[sub]] = (h16)v;
        }
    }
}

// ---------------------------------------------------------------------------
// head: xs (256,128,8) -> per-q Dense(8->32)+ELU+BN+Dense(32->1). Tiny.
// ---------------------------------------------------------------------------
__global__ void fp_head(const h16* __restrict__ act,   // (256, 1024) f16
                        const float* __restrict__ w1,  // (128,8,32)
                        const float* __restrict__ b1,  // (128,32)
                        const float* __restrict__ gamma,
                        const float* __restrict__ beta,
                        const float* __restrict__ mean,
                        const float* __restrict__ var,
                        const float* __restrict__ w2,  // (128,32,1)
                        const float* __restrict__ b2,  // (128,1)
                        float* __restrict__ out)       // (256,128)
{
    __shared__ float sw1[256], ssc[32], ssh[32], sw2[32], sb1[32];
    __shared__ float sb2;
    const int q   = blockIdx.x;
    const int tid = threadIdx.x;
    sw1[tid]       = w1[q * 256 + tid];
    sw1[tid + 128] = w1[q * 256 + 128 + tid];
    if (tid < 32) {
        float scv = gamma[q * 32 + tid] * rsqrtf(var[q * 32 + tid] + FP_EPS);
        ssc[tid] = scv;
        ssh[tid] = beta[q * 32 + tid] - mean[q * 32 + tid] * scv;
        sb1[tid] = b1[q * 32 + tid];
        sw2[tid] = w2[q * 32 + tid];
    }
    if (tid == 0) sb2 = b2[q];
    __syncthreads();

    const int b = blockIdx.y * 128 + tid;
    float xs[8];
#pragma unroll
    for (int si = 0; si < 8; ++si) xs[si] = (float)act[b * 1024 + q * 8 + si];
    float accv = sb2;
#pragma unroll
    for (int u = 0; u < 32; ++u) {
        float h = sb1[u];
#pragma unroll
        for (int si = 0; si < 8; ++si) h += xs[si] * sw1[si * 32 + u];
        h = elu1(h) * ssc[u] + ssh[u];
        accv += h * sw2[u];
    }
    out[b * 128 + q] = accv;
}

// ---------------------------------------------------------------------------
// host
// d_in layout (dict order): [0]=x ; per layer i (base=1+12i): w1,b1,w2,b2,
// bn1{g,b,m,v}, bn2{g,b,m,v} ; head at 97..104: w1,b1,gamma,beta,mean,var,w2,b2
// ---------------------------------------------------------------------------
extern "C" void kernel_launch(void* const* d_in, const int* in_sizes, int n_in,
                              void* d_out, int out_size, void* d_ws, size_t ws_size,
                              hipStream_t stream)
{
    (void)in_sizes; (void)n_in; (void)out_size; (void)ws_size;
    static const int CHt[8] = {64, 64, 128, 128, 256, 256, 512, 512};

    int Cin_[16], Cout_[16], P_[16], Lin_[16], Lout_[16], S_[16], padL_[16];
    int inXs_[16], inPs_[16], inBs_[16], outXs_[16], outPs_[16], outBs_[16];
    int widx[16], bidx[16], gidx[16];

    int H = 256, W = 32, C = 1;
    for (int i = 0; i < 8; ++i) {
        int base = 1 + i * 12;
        // conv1: 1x3, taps along W, stride (1, s1)
        int c  = 2 * i;
        int s1 = (i & 1) ? 2 : 1;
        int Wout = (W + s1 - 1) / s1;
        Cin_[c] = C; Cout_[c] = CHt[i]; P_[c] = H; Lin_[c] = W; Lout_[c] = Wout; S_[c] = s1;
        { int pt = (Wout - 1) * s1 + 3 - W; padL_[c] = pt > 0 ? pt / 2 : 0; }
        inXs_[c]  = C;       inPs_[c]  = W * C;         inBs_[c]  = H * W * C;
        outXs_[c] = CHt[i];  outPs_[c] = Wout * CHt[i]; outBs_[c] = H * Wout * CHt[i];
        widx[c] = base + 0; bidx[c] = base + 1; gidx[c] = base + 4;
        W = Wout; C = CHt[i];

        // conv2: 3x1, taps along H, stride (2, 1)
        c = 2 * i + 1;
        int Hout = (H + 1) / 2;
        Cin_[c] = C; Cout_[c] = C; P_[c] = W; Lin_[c] = H; Lout_[c] = Hout; S_[c] = 2;
        { int pt = (Hout - 1) * 2 + 3 - H; padL_[c] = pt > 0 ? pt / 2 : 0; }
        inXs_[c]  = W * C; inPs_[c]  = C; inBs_[c]  = H * W * C;
        outXs_[c] = W * C; outPs_[c] = C; outBs_[c] = Hout * W * C;
        widx[c] = base + 2; bidx[c] = base + 3; gidx[c] = base + 8;
        H = Hout;
    }

    char*  ws  = (char*)d_ws;
    size_t off = 0;
    auto alloc = [&](size_t bytes) -> size_t {
        off = (off + 255) & ~(size_t)255;
        size_t o = off; off += bytes; return o;
    };
    const size_t actA = alloc((size_t)256 * 256 * 32 * 64 * 2); // conv1 mids (max: layer0)
    const size_t actB = alloc((size_t)256 * 128 * 32 * 64 * 2); // conv2 outs (max: layer0)
    size_t wTo[16], sco[16], sho[16];
    for (int c = 0; c < 16; ++c) {
        wTo[c] = alloc((size_t)3 * Cin_[c] * Cout_[c] * 2);
        sco[c] = alloc((size_t)Cout_[c] * 4);
        sho[c] = alloc((size_t)Cout_[c] * 4);
    }

    // weight prep + BN fold for all 16 convs
    for (int c = 0; c < 16; ++c) {
        int n = 3 * Cin_[c] * Cout_[c];
        fp_prep_conv<<<(n + 255) / 256, 256, 0, stream>>>(
            (const float*)d_in[widx[c]],
            (const float*)d_in[gidx[c] + 0], (const float*)d_in[gidx[c] + 1],
            (const float*)d_in[gidx[c] + 2], (const float*)d_in[gidx[c] + 3],
            (h16*)(ws + wTo[c]), (float*)(ws + sco[c]), (float*)(ws + sho[c]),
            Cin_[c], Cout_[c], n);
    }

    // layer 0 conv1 (Cin=1)
    fp_conv0<<<(256 * 256 * 32) / 256, 256, 0, stream>>>(
        (const float*)d_in[0], (const h16*)(ws + wTo[0]),
        (const float*)d_in[bidx[0]], (const float*)(ws + sco[0]), (const float*)(ws + sho[0]),
        (h16*)(ws + actA));

    // remaining 15 convs: WMMA implicit GEMM; conv1->actA, conv2->actB
    for (int c = 1; c < 16; ++c) {
        const h16* inp = (c & 1) ? (const h16*)(ws + actA) : (const h16*)(ws + actB);
        h16* outp      = (c & 1) ? (h16*)(ws + actB)       : (h16*)(ws + actA);
        int M = 256 * P_[c] * Lout_[c];
        if (Cout_[c] >= 128) {
            dim3 g(M / 64, Cout_[c] / 128);
            fp_conv_wmma<4><<<g, 256, 0, stream>>>(
                inp, (const h16*)(ws + wTo[c]),
                (const float*)d_in[bidx[c]], (const float*)(ws + sco[c]), (const float*)(ws + sho[c]),
                outp,
                Cin_[c], Cout_[c], Lin_[c], Lout_[c], P_[c], S_[c], padL_[c],
                inXs_[c], inPs_[c], inBs_[c], outXs_[c], outPs_[c], outBs_[c]);
        } else {
            dim3 g(M / 64, Cout_[c] / 64);
            fp_conv_wmma<2><<<g, 256, 0, stream>>>(
                inp, (const h16*)(ws + wTo[c]),
                (const float*)d_in[bidx[c]], (const float*)(ws + sco[c]), (const float*)(ws + sho[c]),
                outp,
                Cin_[c], Cout_[c], Lin_[c], Lout_[c], P_[c], S_[c], padL_[c],
                inXs_[c], inPs_[c], inBs_[c], outXs_[c], outPs_[c], outBs_[c]);
        }
    }

    // head on final activations (in actB, (256,1024) f16)
    dim3 gh(128, 2);
    fp_head<<<gh, 128, 0, stream>>>(
        (const h16*)(ws + actB),
        (const float*)d_in[97], (const float*)d_in[98],
        (const float*)d_in[99], (const float*)d_in[100],
        (const float*)d_in[101], (const float*)d_in[102],
        (const float*)d_in[103], (const float*)d_in[104],
        (float*)d_out);
}
